// GAT_55293408968749
// MI455X (gfx1250) — matrix operand order repaired and checked by reading, not compiled, not run on hardware
//
#include <hip/hip_runtime.h>

// ---------- types ----------
typedef __attribute__((ext_vector_type(16))) __bf16 bf16x16;
typedef __attribute__((ext_vector_type(8)))  __bf16 bf16x8;
typedef __attribute__((ext_vector_type(8)))  float  f32x8;

#define NEG_SLOPE 0.2f

static __device__ __forceinline__ unsigned short f32_to_bf16(float f) {
    unsigned int u = __float_as_uint(f);
    if ((u & 0x7FFFFFFFu) > 0x7F800000u) return (unsigned short)0x7FC0; // NaN
    unsigned int r = u + 0x7FFFu + ((u >> 16) & 1u);                    // RNE
    return (unsigned short)(r >> 16);
}

static __device__ __forceinline__ void atomicMaxF(float* addr, float val) {
    if (val >= 0.0f) atomicMax((int*)addr, __float_as_int(val));
    else             atomicMin((unsigned int*)addr, __float_as_uint(val));
}

// ---------- conversion kernels ----------
__global__ void k_f32_to_bf16(const float* __restrict__ x,
                              unsigned short* __restrict__ y, long long n) {
    long long t = (long long)blockIdx.x * blockDim.x + threadIdx.x;
    if (t < n) y[t] = f32_to_bf16(x[t]);
}

// W [K, Nout] f32  ->  Wt [Nout, K] bf16
__global__ void k_w_transpose_bf16(const float* __restrict__ W,
                                   unsigned short* __restrict__ Wt,
                                   int K, int Nout) {
    long long t = (long long)blockIdx.x * blockDim.x + threadIdx.x;
    if (t >= (long long)K * Nout) return;
    int k = (int)(t % K);
    int n = (int)(t / K);
    Wt[(size_t)n * K + k] = f32_to_bf16(W[(size_t)k * Nout + n]);
}

// ---------- WMMA fragment loaders (wave32 VGPR layouts, CDNA5 ISA 7.12.2) ----------
// A 16x32 bf16: lane holds row (lane&15); K = (lane>>4)*8 + [0..7] and +16..+23
static __device__ __forceinline__ bf16x16 load_frag_a(const __bf16* __restrict__ p) {
    bf16x8 lo = *(const bf16x8*)p;
    bf16x8 hi = *(const bf16x8*)(p + 16);
    bf16x16 r;
#pragma unroll
    for (int i = 0; i < 8; ++i) { r[i] = lo[i]; r[i + 8] = hi[i]; }
    return r;
}
// B 32x16 bf16 (from transposed W): lane holds col (lane&15); K = (lane>>4)*8 + [0..7] and +16..+23
static __device__ __forceinline__ bf16x16 load_frag_b(const __bf16* __restrict__ p) {
    bf16x8 lo = *(const bf16x8*)p;
    bf16x8 hi = *(const bf16x8*)(p + 16);
    bf16x16 r;
#pragma unroll
    for (int i = 0; i < 8; ++i) { r[i] = lo[i]; r[i + 8] = hi[i]; }
    return r;
}

// ---------- WMMA GEMM: C[M,N] f32 = A[M,K] bf16 * Bt[N,K] bf16 ----------
// One wave computes a 32x64 block: 2 M-subtiles x 4 N-subtiles, 8 WMMAs per k-step,
// 12 b128 loads per k-step (1.5 loads/WMMA). TWO_M=false handles ragged last M tile.
template <bool TWO_M>
static __device__ __forceinline__ void gemm_body(const __bf16* __restrict__ A,
                                                 const __bf16* __restrict__ Bt,
                                                 float* __restrict__ C,
                                                 int tm, int tn, int N, int K, int lane) {
    const int half = lane >> 4;
    const int mrow = lane & 15;
    const __bf16* Arow0 = A + (size_t)(tm + mrow) * K + half * 8;
    const __bf16* Arow1 = Arow0 + (size_t)16 * K;
    const __bf16* B0    = Bt + (size_t)(tn + mrow) * K + half * 8;

    f32x8 c0[4] = {};
    f32x8 c1[4] = {};
    for (int k = 0; k < K; k += 32) {
        __builtin_prefetch(Arow0 + k + 256, 0, 3);   // global_prefetch_b8
        bf16x16 a0 = load_frag_a(Arow0 + k);
        bf16x16 a1;
        if (TWO_M) a1 = load_frag_a(Arow1 + k);
#pragma unroll
        for (int j = 0; j < 4; ++j) {
            bf16x16 b = load_frag_b(B0 + (size_t)(16 * j) * K + k);
            c0[j] = __builtin_amdgcn_wmma_f32_16x16x32_bf16(
                        false, a0, false, b, (short)0, c0[j], false, false);
            if (TWO_M)
                c1[j] = __builtin_amdgcn_wmma_f32_16x16x32_bf16(
                            false, a1, false, b, (short)0, c1[j], false, false);
        }
    }
    // D layout: VGPR i -> row i + 8*(lane>>4), col = lane&15
#pragma unroll
    for (int j = 0; j < 4; ++j) {
#pragma unroll
        for (int i = 0; i < 8; ++i) {
            C[(size_t)(tm + i + half * 8) * N + tn + 16 * j + mrow] = c0[j][i];
            if (TWO_M)
                C[(size_t)(tm + 16 + i + half * 8) * N + tn + 16 * j + mrow] = c1[j][i];
        }
    }
}

__global__ void k_gemm_wmma(const unsigned short* __restrict__ Au,
                            const unsigned short* __restrict__ Btu,
                            float* __restrict__ C, int M, int N, int K) {
    int wave = (int)(((long long)blockIdx.x * blockDim.x + threadIdx.x) >> 5);
    int lane = threadIdx.x & 31;
    int ntiles = N >> 6;                 // 64-wide tiles
    int mtiles = (M + 31) >> 5;          // 32-high tiles
    if (wave >= mtiles * ntiles) return; // wave-uniform: EXEC stays full
    int tm = (wave / ntiles) << 5;
    int tn = (wave % ntiles) << 6;
    const __bf16* A  = (const __bf16*)Au;
    const __bf16* Bt = (const __bf16*)Btu;
    if (tm + 16 < M) gemm_body<true >(A, Bt, C, tm, tn, N, K, lane);
    else             gemm_body<false>(A, Bt, C, tm, tn, N, K, lane);
}

// ---------- attention scores: el/er [N*H] ----------
__global__ void k_attn_scores(const float* __restrict__ ft,
                              const float* __restrict__ al,
                              const float* __restrict__ ar,
                              float* __restrict__ el, float* __restrict__ er,
                              int NH, int H, int F) {
    int w = (int)(((long long)blockIdx.x * blockDim.x + threadIdx.x) >> 5);
    int lane = threadIdx.x & 31;
    if (w >= NH) return;
    int h = w % H;
    const float* row = ft + (size_t)w * F;
    float sl = 0.f, sr = 0.f;
    for (int f = lane; f < F; f += 32) {
        float v = row[f];
        sl += v * al[h * F + f];
        sr += v * ar[h * F + f];
    }
#pragma unroll
    for (int m = 16; m; m >>= 1) {
        sl += __shfl_xor(sl, m, 32);
        sr += __shfl_xor(sr, m, 32);
    }
    if (lane == 0) { el[w] = sl; er[w] = sr; }
}

// ---------- init kernels ----------
__global__ void k_init_softmax(float* __restrict__ maxb, float* __restrict__ sumb, int n) {
    int t = blockIdx.x * blockDim.x + threadIdx.x;
    if (t < n) { maxb[t] = __int_as_float(0xFF800000); sumb[t] = 0.f; }
}
__global__ void k_init_rst(float* __restrict__ rst, const float* __restrict__ b,
                           long long total, int HF) {
    long long t = (long long)blockIdx.x * blockDim.x + threadIdx.x;
    if (t < total) rst[t] = b[(int)(t % HF)];
}

// ---------- edge kernels ----------
__global__ void k_edge_score(const int* __restrict__ src, const int* __restrict__ dst,
                             const float* __restrict__ el, const float* __restrict__ er,
                             float* __restrict__ esc, float* __restrict__ maxb,
                             int E, int H) {
    int t = blockIdx.x * blockDim.x + threadIdx.x;
    if (t >= E * H) return;
    int e = t / H, h = t % H;
    float s = el[src[e] * H + h] + er[dst[e] * H + h];
    s = (s > 0.f) ? s : NEG_SLOPE * s;              // leaky_relu
    esc[t] = s;
    atomicMaxF(&maxb[dst[e] * H + h], s);
}

__global__ void k_edge_exp(const int* __restrict__ dst,
                           float* __restrict__ esc, const float* __restrict__ maxb,
                           float* __restrict__ sumb, int E, int H) {
    int t = blockIdx.x * blockDim.x + threadIdx.x;
    if (t >= E * H) return;
    int e = t / H, h = t % H;
    float m = maxb[dst[e] * H + h];
    if (!isfinite(m)) m = 0.f;                      // zero-in-degree safety
    float ex = expf(esc[t] - m);
    esc[t] = ex;
    atomicAdd(&sumb[dst[e] * H + h], ex);
}

__global__ void k_edge_aggregate(const int* __restrict__ src, const int* __restrict__ dst,
                                 const float* __restrict__ esc, const float* __restrict__ sumb,
                                 const float* __restrict__ ft, float* __restrict__ rst,
                                 long long total, int H, int F, int HF) {
    long long t = (long long)blockIdx.x * blockDim.x + threadIdx.x;
    if (t >= total) return;
    int hf = (int)(t % HF);
    int e  = (int)(t / HF);
    int h  = hf / F;
    int s  = src[e], d = dst[e];
    float a = esc[e * H + h] / sumb[d * H + h];
    atomicAdd(&rst[(size_t)d * HF + hf], ft[(size_t)s * HF + hf] * a);
}

// ---------- finalize ----------
__global__ void k_elu_to_bf16(const float* __restrict__ rst,
                              unsigned short* __restrict__ hb, long long n) {
    long long t = (long long)blockIdx.x * blockDim.x + threadIdx.x;
    if (t >= n) return;
    float x = rst[t];
    x = (x > 0.f) ? x : expm1f(x);                  // elu
    hb[t] = f32_to_bf16(x);
}

__global__ void k_mean_heads(const float* __restrict__ rst, float* __restrict__ out,
                             long long total, int H, int F) {
    long long t = (long long)blockIdx.x * blockDim.x + threadIdx.x;
    if (t >= total) return;
    int f = (int)(t % F);
    long long n = t / F;
    float s = 0.f;
    for (int h = 0; h < H; ++h) s += rst[n * (size_t)(H * F) + h * F + f];
    out[t] = s / (float)H;
}

// ---------- host orchestration ----------
static inline int cdiv_i(long long a, int b) { return (int)((a + b - 1) / b); }

static void run_gat_layer(const unsigned short* hb_in, int Din,
                          const float* W, const float* al, const float* ar, const float* b,
                          const int* src, const int* dst, int E, int Nn, int H, int F,
                          unsigned short* Wt, float* ft, float* rst,
                          float* el, float* er, float* maxb, float* sumb, float* esc,
                          unsigned short* hb_out /*flatten path*/, float* f32_out /*mean path*/,
                          hipStream_t stream) {
    const int HF = H * F;
    // 1) Wt = bf16(W^T)
    k_w_transpose_bf16<<<cdiv_i((long long)Din * HF, 256), 256, 0, stream>>>(W, Wt, Din, HF);
    // 2) ft = hb_in @ W  (WMMA, 32x64 blocks per wave)
    {
        long long waves = (long long)((Nn + 31) >> 5) * (HF >> 6);
        k_gemm_wmma<<<cdiv_i(waves * 32, 256), 256, 0, stream>>>(hb_in, Wt, ft, Nn, HF, Din);
    }
    // 3) el/er
    k_attn_scores<<<cdiv_i((long long)Nn * H * 32, 256), 256, 0, stream>>>(
        ft, al, ar, el, er, Nn * H, H, F);
    // 4) init softmax buffers + rst = bias
    k_init_softmax<<<cdiv_i((long long)Nn * H, 256), 256, 0, stream>>>(maxb, sumb, Nn * H);
    k_init_rst<<<cdiv_i((long long)Nn * HF, 256), 256, 0, stream>>>(rst, b, (long long)Nn * HF, HF);
    // 5) edge softmax
    k_edge_score<<<cdiv_i((long long)E * H, 256), 256, 0, stream>>>(src, dst, el, er, esc, maxb, E, H);
    k_edge_exp<<<cdiv_i((long long)E * H, 256), 256, 0, stream>>>(dst, esc, maxb, sumb, E, H);
    // 6) weighted aggregation
    k_edge_aggregate<<<cdiv_i((long long)E * HF, 256), 256, 0, stream>>>(
        src, dst, esc, sumb, ft, rst, (long long)E * HF, H, F, HF);
    // 7) finalize
    if (hb_out)
        k_elu_to_bf16<<<cdiv_i((long long)Nn * HF, 256), 256, 0, stream>>>(rst, hb_out, (long long)Nn * HF);
    else
        k_mean_heads<<<cdiv_i((long long)Nn * F, 256), 256, 0, stream>>>(rst, f32_out, (long long)Nn * F, H, F);
}

extern "C" void kernel_launch(void* const* d_in, const int* in_sizes, int n_in,
                              void* d_out, int out_size, void* d_ws, size_t ws_size,
                              hipStream_t stream) {
    const float* feat = (const float*)d_in[0];
    const int*   src  = (const int*)d_in[1];
    const int*   dst  = (const int*)d_in[2];
    const float* W1  = (const float*)d_in[3];
    const float* al1 = (const float*)d_in[4];
    const float* ar1 = (const float*)d_in[5];
    const float* b1  = (const float*)d_in[6];
    const float* W2  = (const float*)d_in[7];
    const float* al2 = (const float*)d_in[8];
    const float* ar2 = (const float*)d_in[9];
    const float* b2  = (const float*)d_in[10];
    const float* W3  = (const float*)d_in[11];
    const float* al3 = (const float*)d_in[12];
    const float* ar3 = (const float*)d_in[13];
    const float* b3  = (const float*)d_in[14];

    const int Nn = in_sizes[0] / 256;   // 10000
    const int E  = in_sizes[1];         // 160000
    const int HMAX = 6, DMAX = 1024;

    char* ws = (char*)d_ws;
    size_t off = 0;
    auto alloc = [&](size_t bytes) -> char* {
        char* p = ws + off;
        off = (off + bytes + 255) & ~(size_t)255;
        return p;
    };
    unsigned short* bufA = (unsigned short*)alloc((size_t)Nn * DMAX * 2);
    unsigned short* bufB = (unsigned short*)alloc((size_t)Nn * DMAX * 2);
    unsigned short* Wt   = (unsigned short*)alloc((size_t)DMAX * DMAX * 2);
    float* ft   = (float*)alloc((size_t)Nn * DMAX * 4);
    float* rst  = (float*)alloc((size_t)Nn * DMAX * 4);
    float* el   = (float*)alloc((size_t)Nn * HMAX * 4);
    float* er   = (float*)alloc((size_t)Nn * HMAX * 4);
    float* maxb = (float*)alloc((size_t)Nn * HMAX * 4);
    float* sumb = (float*)alloc((size_t)Nn * HMAX * 4);
    float* esc  = (float*)alloc((size_t)E * HMAX * 4);
    (void)ws_size; (void)n_in; (void)out_size;

    // layer 1 input: bf16(feat) -> bufA
    k_f32_to_bf16<<<cdiv_i((long long)Nn * 256, 256), 256, 0, stream>>>(
        feat, bufA, (long long)Nn * 256);

    // layer 1: Din=256, H=4, F=256, flatten -> bufB
    run_gat_layer(bufA, 256, W1, al1, ar1, b1, src, dst, E, Nn, 4, 256,
                  Wt, ft, rst, el, er, maxb, sumb, esc, bufB, nullptr, stream);
    // layer 2: Din=1024, H=4, F=256, flatten -> bufA
    run_gat_layer(bufB, 1024, W2, al2, ar2, b2, src, dst, E, Nn, 4, 256,
                  Wt, ft, rst, el, er, maxb, sumb, esc, bufA, nullptr, stream);
    // layer 3: Din=1024, H=6, F=64, mean -> d_out
    run_gat_layer(bufA, 1024, W3, al3, ar3, b3, src, dst, E, Nn, 6, 64,
                  Wt, ft, rst, el, er, maxb, sumb, esc, nullptr, (float*)d_out, stream);
}
